// DCAGCell_9895604650326
// MI455X (gfx1250) — matrix operand order repaired and checked by reading, not compiled, not run on hardware
//
#include <hip/hip_runtime.h>
#include <stdint.h>
#include <stddef.h>

// ---------------------------------------------------------------------------
// DCAGCell for MI455X (gfx1250, wave32, WMMA).
// Heavy GEMMs on v_wmma_f32_16x16x32_bf16 (f32 accum), 64-deep K tiles
// (8 WMMA per barrier interval). Dominant M@P GEMM uses deterministic
// split-K x4 into partial buffers (summed in the gate kernels) to raise the
// grid from 160 -> 640 workgroups. M (bf16, 52MB) stays L2-resident (192MB).
// Algebraic refactor: lin(M @ x, W) == M @ (x @ W^T) (bias after) cuts the
// layer-1 mixing GEMM K-cost 4x. Requires ws_size >= ~270 MB.
// ---------------------------------------------------------------------------

typedef __attribute__((ext_vector_type(16))) __bf16 v16bf;
typedef __attribute__((ext_vector_type(8)))  float  v8f;

struct __align__(16) U4 { unsigned int a, b, c, d; };
union FragBF { U4 u[2]; v16bf v; };

constexpr int kB  = 4, kN = 5000, kS = 256, kD = 64, kH = 3, kR = 100;
constexpr int kNT = kB * kN;          // 20000 flattened nodes
constexpr int kE  = 320000;
constexpr int kET = kE + kNT;         // edges + self loops
constexpr int kHD = kH * kD;          // 192
constexpr int kNP  = 5120;            // N padded to mult of 128 (GEMM M and K dim)
constexpr int kRP  = 128;             // R=100 padded
constexpr int kNTP = 20096;           // NT padded to mult of 128
constexpr int kKS  = 4;               // split-K factor for the M@P GEMM
constexpr int kTK  = 64;              // K depth of one LDS tile

// ---- workspace layout ------------------------------------------------------
// bf16 (ushort) staging region (zero-filled each launch so pads are zero)
constexpr size_t U_MBF  = 0;                                   // M bf16 [kNP][kNP]
constexpr size_t U_TS   = U_MBF  + (size_t)kNP * kNP;          // Ts  [kNP][kRP]
constexpr size_t U_TT   = U_TS   + (size_t)kNP * kRP;          // Tt  [kRP][kNP]
constexpr size_t U_X1   = U_TT   + (size_t)kRP * kNP;          // x   [kNTP][kS]
constexpr size_t U_X23  = U_X1   + (size_t)kNTP * kS;          // t1/r2 [kNTP][kD]
constexpr size_t U_PB   = U_X23  + (size_t)kNTP * kD;          // P per-batch B [kB][kNP][kD]
constexpr size_t U_WOT  = U_PB   + (size_t)kB * kNP * kD;      // Wo^T  [kS][kD]
constexpr size_t U_WL1T = U_WOT  + (size_t)kS * kD;            // Wl1^T [kS][kD]
constexpr size_t U_WL2T = U_WL1T + (size_t)kS * kD;            // Wl2^T [kD][kD]
constexpr size_t U_WL3T = U_WL2T + (size_t)kD * kD;            // Wl3^T [kD][kD]
constexpr size_t U_WG1T = U_WL3T + (size_t)kD * kD;            // Wg1^T [kS][kHD]
constexpr size_t U_WG2T = U_WG1T + (size_t)kS * kHD;           // Wg2^T [kD][kHD]
constexpr size_t U_WG3T = U_WG2T + (size_t)kD * kHD;           // Wg3^T [kD][kHD]
constexpr size_t U_END  = U_WG3T + (size_t)kD * kHD;
constexpr size_t FBYTE  = ((U_END * 2 + 255) / 256) * 256;     // float region base (bytes)
// float region (offsets in floats from FBYTE)
constexpr size_t F_LOG  = 0;                                   // logits [kNP][kNP]
constexpr size_t F_P    = F_LOG  + (size_t)kNP * kNP;          // P = X@Wl^T [kNTP][kD]
constexpr size_t F_ORG  = F_P    + (size_t)kNTP * kD;          // origin [kNTP][kD]
constexpr size_t F_ADP  = F_ORG  + (size_t)kNTP * kD;          // M@P partials [kB*kKS][kNP][kD]
constexpr size_t F_H    = F_ADP  + (size_t)kB * kKS * kNP * kD;// h [kNTP][kHD]
constexpr size_t F_ALS  = F_H    + (size_t)kNTP * kHD;         // al_src [kNT][kH]
constexpr size_t F_ALD  = F_ALS  + (size_t)kNT * kH;           // al_dst [kNT][kH]
constexpr size_t F_EMAX = F_ALD  + (size_t)kNT * kH;           // seg-max (ordered u32)
constexpr size_t F_DEN  = F_EMAX + (size_t)kNT * kH;           // seg-sum of exp
constexpr size_t F_EX   = F_DEN  + (size_t)kNT * kH;           // exp per edge [kET][kH]
constexpr size_t F_GACC = F_EX   + (size_t)kET * kH;           // gat accum [kNT][kHD]
constexpr size_t F_OUT1 = F_GACC + (size_t)kNT * kHD;
constexpr size_t F_OUT2 = F_OUT1 + (size_t)kNT * kD;
constexpr size_t F_XN   = F_OUT2 + (size_t)kNT * kD;           // next-layer input

// ---- small device helpers --------------------------------------------------
__device__ __forceinline__ unsigned short f2bf(float f) {
  unsigned u = __float_as_uint(f);
  unsigned r = (u + 0x7FFFu + ((u >> 16) & 1u)) >> 16;   // round-to-nearest-even
  return (unsigned short)r;
}
__device__ __forceinline__ unsigned encf(float f) {      // order-preserving f32->u32
  unsigned u = __float_as_uint(f);
  return (u & 0x80000000u) ? ~u : (u | 0x80000000u);
}
__device__ __forceinline__ float decf(unsigned e) {
  return __uint_as_float((e & 0x80000000u) ? (e & 0x7FFFFFFFu) : ~e);
}

// ---- utility kernels -------------------------------------------------------
__global__ void k_fill_u32(unsigned* p, size_t n, unsigned v) {
  size_t i  = (size_t)blockIdx.x * blockDim.x + threadIdx.x;
  size_t st = (size_t)gridDim.x * blockDim.x;
  for (; i < n; i += st) p[i] = v;
}

// f32 [rows][cols] (contiguous) -> bf16 [rows][ldd] (pads pre-zeroed)
__global__ void k_cvt_pad(const float* __restrict__ src, unsigned short* __restrict__ dst,
                          int rows, int cols, int ldd) {
  size_t n  = (size_t)rows * cols;
  size_t i  = (size_t)blockIdx.x * blockDim.x + threadIdx.x;
  size_t st = (size_t)gridDim.x * blockDim.x;
  for (; i < n; i += st) {
    int r = (int)(i / cols), c = (int)(i - (size_t)r * cols);
    dst[(size_t)r * ldd + c] = f2bf(src[i]);
  }
}

// W f32 [dout][k] -> W^T bf16 [k][dout]
__global__ void k_cvt_T(const float* __restrict__ W, unsigned short* __restrict__ WT,
                        int dout, int k) {
  size_t n  = (size_t)dout * k;
  size_t i  = (size_t)blockIdx.x * blockDim.x + threadIdx.x;
  size_t st = (size_t)gridDim.x * blockDim.x;
  for (; i < n; i += st) {
    int o = (int)(i / k), kk = (int)(i - (size_t)o * k);
    WT[(size_t)kk * dout + o] = f2bf(W[i]);
  }
}

// P f32 [kNT][kD] -> per-batch B-matrix bf16 [kB][kNP][kD] (pad rows pre-zeroed)
__global__ void k_cvt_pb(const float* __restrict__ P, unsigned short* __restrict__ Pb) {
  int t = blockIdx.x * blockDim.x + threadIdx.x;
  if (t >= kNT * kD) return;
  int n = t >> 6, d = t & 63;
  int b = n / kN, m = n - b * kN;
  Pb[((size_t)b * kNP + m) * kD + d] = f2bf(P[t]);
}

// ---- generic bf16 WMMA GEMM: C[M x N] = A[M x K] * B[K x N] ---------------
// block = 128(M) x 64(N), 256 threads = 8 wave32 waves in 4x2; wave = 32x32.
// 64-deep K tiles -> 8 v_wmma_f32_16x16x32_bf16 per barrier interval.
// Optional split-K: grid.z = batch*ksplit + kslice; each slice writes its own
// C plane (summed later) so the launch stays deterministic (no float atomics).
__global__ __launch_bounds__(256)
void k_gemm_bf16(const unsigned short* __restrict__ A, int lda, long long strideA,
                 const unsigned short* __restrict__ Bm, int ldb, long long strideB,
                 float* __restrict__ C, int ldc, long long strideC, int K, int ksplit) {
  __shared__ __align__(16) unsigned short sA[128 * kTK];  // [row][k]
  __shared__ __align__(16) unsigned short sB[64 * kTK];   // transposed: [col][k]

  const int tid  = threadIdx.x;
  const int lane = tid & 31;
  const int wave = tid >> 5;
  const int wm   = wave >> 1;           // 0..3
  const int wn   = wave & 1;            // 0..1
  const long long rowBase = (long long)blockIdx.y * 128;
  const long long colBase = (long long)blockIdx.x * 64;

  const int zb = blockIdx.z / ksplit;   // batch
  const int zk = blockIdx.z - zb * ksplit;
  const int Kper = K / ksplit;
  const int kbeg = zk * Kper, kend = kbeg + Kper;

  A  += (long long)zb * strideA;
  Bm += (long long)zb * strideB;
  C  += (long long)blockIdx.z * strideC;

  v8f acc[2][2] = {};

  const int arow = tid >> 1;            // 0..127 (A tile row)
  const int aseg = (tid & 1) * 32;      // 32-elem segment in k
  const int bkr  = tid >> 2;            // 0..63  (B tile k-row)
  const int bn   = (tid & 3) * 16;      // 16-elem segment in n

  for (int k0 = kbeg; k0 < kend; k0 += kTK) {
    // ---- global -> registers: issue ALL tile loads first (one clause, many
    // outstanding LOADcnt) so latency overlaps; then drain into LDS. --------
    const unsigned short* ag = A + (rowBase + arow) * lda + k0 + aseg;
    const unsigned short* bg = Bm + (long long)(k0 + bkr) * ldb + colBase + bn;
    U4 at0 = reinterpret_cast<const U4*>(ag)[0];
    U4 at1 = reinterpret_cast<const U4*>(ag)[1];
    U4 at2 = reinterpret_cast<const U4*>(ag)[2];
    U4 at3 = reinterpret_cast<const U4*>(ag)[3];
    U4 bt0 = reinterpret_cast<const U4*>(bg)[0];
    U4 bt1 = reinterpret_cast<const U4*>(bg)[1];

    if (k0 + kTK < kend) {  // speculative prefetch of next K-tile
      __builtin_prefetch(ag + kTK, 0, 1);
      __builtin_prefetch(Bm + (long long)(k0 + kTK + bkr) * ldb + colBase + bn, 0, 1);
    }

    // registers -> LDS (A row-major [row][k]; B transposed [col][k])
    *reinterpret_cast<U4*>(&sA[arow * kTK + aseg])      = at0;
    *reinterpret_cast<U4*>(&sA[arow * kTK + aseg + 8])  = at1;
    *reinterpret_cast<U4*>(&sA[arow * kTK + aseg + 16]) = at2;
    *reinterpret_cast<U4*>(&sA[arow * kTK + aseg + 24]) = at3;
    const unsigned short* bs0 = reinterpret_cast<const unsigned short*>(&bt0);
    const unsigned short* bs1 = reinterpret_cast<const unsigned short*>(&bt1);
#pragma unroll
    for (int j = 0; j < 8; ++j) {
      sB[(bn + j) * kTK + bkr]     = bs0[j];
      sB[(bn + 8 + j) * kTK + bkr] = bs1[j];
    }
    __syncthreads();

    // A frag per-lane layout (ISA 7.12.2, 16-bit A 16x32):
    //   lanes 0-15: K 0..7 then 16..23 ; lanes 16-31: K 8..15 then 24..31
    const int kc0 = (lane >> 4) * 8;
    const int kb  = (lane >> 4) * 8;
#pragma unroll
    for (int ks = 0; ks < kTK; ks += 32) {
      FragBF fa[2], fb[2];
#pragma unroll
      for (int mt = 0; mt < 2; ++mt) {
        int r = wm * 32 + mt * 16 + (lane & 15);
        fa[mt].u[0] = *reinterpret_cast<const U4*>(&sA[r * kTK + ks + kc0]);
        fa[mt].u[1] = *reinterpret_cast<const U4*>(&sA[r * kTK + ks + kc0 + 16]);
      }
      // B frag: lane c (0-15) = column c, K 0..7 then 16..23 ; lane 16+c = column c, K 8..15 then 24..31
#pragma unroll
      for (int nt = 0; nt < 2; ++nt) {
        int c = wn * 32 + nt * 16 + (lane & 15);
        fb[nt].u[0] = *reinterpret_cast<const U4*>(&sB[c * kTK + ks + kb]);
        fb[nt].u[1] = *reinterpret_cast<const U4*>(&sB[c * kTK + ks + kb + 16]);
      }
#pragma unroll
      for (int mt = 0; mt < 2; ++mt)
#pragma unroll
        for (int nt = 0; nt < 2; ++nt)
          acc[mt][nt] = __builtin_amdgcn_wmma_f32_16x16x32_bf16(
              false, fa[mt].v, false, fb[nt].v, (short)0, acc[mt][nt], false, false);
    }
    __syncthreads();
  }

  // C frag layout: VGPR v, lanes 0-15 -> row v, lanes 16-31 -> row v+8; col = lane&15
#pragma unroll
  for (int mt = 0; mt < 2; ++mt)
#pragma unroll
    for (int nt = 0; nt < 2; ++nt) {
      long long col = colBase + wn * 32 + nt * 16 + (lane & 15);
      long long r0  = rowBase + wm * 32 + mt * 16 + (lane >> 4) * 8;
#pragma unroll
      for (int v = 0; v < 8; ++v)
        C[(r0 + v) * ldc + col] = acc[mt][nt][v];
    }
}

// ---- row softmax of relu(logits) -> bf16 M --------------------------------
__global__ __launch_bounds__(256)
void k_softmax(const float* __restrict__ L, unsigned short* __restrict__ Mout) {
  __shared__ float red[256];
  const int row = blockIdx.x;
  const int tid = threadIdx.x;
  const float* Lr = L + (size_t)row * kNP;
  float mx = 0.f;                                     // relu => all values >= 0
  for (int c = tid; c < kN; c += 256) { float z = fmaxf(Lr[c], 0.f); mx = fmaxf(mx, z); }
  red[tid] = mx; __syncthreads();
  for (int s = 128; s > 0; s >>= 1) { if (tid < s) red[tid] = fmaxf(red[tid], red[tid + s]); __syncthreads(); }
  mx = red[0]; __syncthreads();
  float sum = 0.f;
  for (int c = tid; c < kN; c += 256) { float z = fmaxf(Lr[c], 0.f); sum += expf(z - mx); }
  red[tid] = sum; __syncthreads();
  for (int s = 128; s > 0; s >>= 1) { if (tid < s) red[tid] += red[tid + s]; __syncthreads(); }
  float inv = 1.f / red[0];
  unsigned short* Mr = Mout + (size_t)row * kNP;
  for (int c = tid; c < kN; c += 256) {
    float z = fmaxf(Lr[c], 0.f);
    Mr[c] = f2bf(expf(z - mx) * inv);
  }
}

// ---- GAT pieces ------------------------------------------------------------
__global__ void k_attn(const float* __restrict__ h, const float* __restrict__ a_s,
                       const float* __restrict__ a_d, float* __restrict__ als,
                       float* __restrict__ ald) {
  int t = blockIdx.x * blockDim.x + threadIdx.x;
  if (t >= kNT * kH) return;
  int n = t / kH, hh = t - n * kH;
  const float* hp = h + (size_t)n * kHD + hh * kD;
  const float* sp = a_s + hh * kD;
  const float* dp = a_d + hh * kD;
  float ss = 0.f, sd = 0.f;
#pragma unroll 8
  for (int d = 0; d < kD; ++d) { float hv = hp[d]; ss += hv * sp[d]; sd += hv * dp[d]; }
  als[t] = ss; ald[t] = sd;
}

__device__ __forceinline__ void edge_sd(const int* ei, int e, int& s, int& d) {
  if (e < kE) { s = ei[e]; d = ei[kE + e]; }
  else        { s = e - kE; d = s; }
}

__global__ void k_edge_max(const int* __restrict__ ei, const float* __restrict__ als,
                           const float* __restrict__ ald, unsigned* __restrict__ emax) {
  int e = blockIdx.x * blockDim.x + threadIdx.x;
  if (e >= kET) return;
  int s, d; edge_sd(ei, e, s, d);
#pragma unroll
  for (int hh = 0; hh < kH; ++hh) {
    float v = als[s * kH + hh] + ald[d * kH + hh];
    v = v > 0.f ? v : 0.2f * v;                     // leaky_relu(0.2)
    atomicMax(&emax[d * kH + hh], encf(v));
  }
}

__global__ void k_edge_exp(const int* __restrict__ ei, const float* __restrict__ als,
                           const float* __restrict__ ald, const unsigned* __restrict__ emax,
                           float* __restrict__ ex, float* __restrict__ den) {
  int e = blockIdx.x * blockDim.x + threadIdx.x;
  if (e >= kET) return;
  int s, d; edge_sd(ei, e, s, d);
#pragma unroll
  for (int hh = 0; hh < kH; ++hh) {
    float v = als[s * kH + hh] + ald[d * kH + hh];
    v = v > 0.f ? v : 0.2f * v;
    float ev = expf(v - decf(emax[d * kH + hh]));
    ex[(size_t)e * kH + hh] = ev;
    atomicAdd(&den[d * kH + hh], ev);
  }
}

__global__ void k_edge_scatter(const int* __restrict__ ei, const float* __restrict__ ex,
                               const float* __restrict__ den, const float* __restrict__ h,
                               float* __restrict__ gacc) {
  int t = blockIdx.x * blockDim.x + threadIdx.x;
  if (t >= kET * kD) return;
  int e = t >> 6, dd = t & 63;
  int s, d; edge_sd(ei, e, s, d);
#pragma unroll
  for (int hh = 0; hh < kH; ++hh) {
    float coef = ex[(size_t)e * kH + hh] / den[d * kH + hh];
    atomicAdd(&gacc[(size_t)d * kHD + hh * kD + dd],
              coef * h[(size_t)s * kHD + hh * kD + dd]);
  }
}

// ---- gated combine kernels -------------------------------------------------
__device__ __forceinline__ void gate_common(const float* gacc, const float* adpP,
                                            const float* bg, const float* bl,
                                            int t, float& gm, float& sg) {
  int n = t >> 6, dd = t & 63;
  gm = (gacc[(size_t)n * kHD + dd] + gacc[(size_t)n * kHD + kD + dd] +
        gacc[(size_t)n * kHD + 2 * kD + dd]) * (1.f / 3.f) + bg[dd];
  int b = n / kN, m = n - b * kN;
  // sum the kKS split-K partial planes (deterministic split-K reduction)
  const float* ap = adpP + (((size_t)b * kKS) * kNP + m) * kD + dd;
  const size_t pl = (size_t)kNP * kD;
  float av = ap[0] + ap[pl] + ap[2 * pl] + ap[3 * pl] + bl[dd];
  sg = 1.f / (1.f + expf(-av));
}

__global__ void k_gate1(const float* gacc, const float* adpP, const float* bg, const float* bl,
                        const float* origin, const float* bo, float* out1, float* xnext) {
  int t = blockIdx.x * blockDim.x + threadIdx.x;
  if (t >= kNT * kD) return;
  float gm, sg; gate_common(gacc, adpP, bg, bl, t, gm, sg);
  float org = origin[t] + bo[t & 63];
  float o = tanhf(gm) * sg + org * (1.f - sg);
  out1[t] = o;
  xnext[t] = tanhf(o);
}

__global__ void k_gate2(const float* gacc, const float* adpP, const float* bg, const float* bl,
                        const float* prev, float* out2, float* xnext) {
  int t = blockIdx.x * blockDim.x + threadIdx.x;
  if (t >= kNT * kD) return;
  float gm, sg; gate_common(gacc, adpP, bg, bl, t, gm, sg);
  float gl = gm > 0.f ? gm : 0.01f * gm;            // leaky_relu(0.01)
  float o = gl * sg + prev[t] * (1.f - sg);
  out2[t] = o;
  xnext[t] = fmaxf(o, 0.f);
}

__global__ void k_gate3(const float* gacc, const float* adpP, const float* bg, const float* bl,
                        const float* prev, float* dout) {
  int t = blockIdx.x * blockDim.x + threadIdx.x;
  if (t >= kNT * kD) return;
  float gm, sg; gate_common(gacc, adpP, bg, bl, t, gm, sg);
  float o = fmaxf(gm, 0.f) * sg + prev[t] * (1.f - sg);
  dout[t] = fmaxf(o, 0.f);
}

// ---------------------------------------------------------------------------
extern "C" void kernel_launch(void* const* d_in, const int* in_sizes, int n_in,
                              void* d_out, int out_size, void* d_ws, size_t ws_size,
                              hipStream_t stream) {
  (void)in_sizes; (void)n_in; (void)out_size; (void)ws_size;

  const float* T_x = (const float*)d_in[0];
  const int*   ei  = (const int*)  d_in[2];
  const float* Ts  = (const float*)d_in[5];
  const float* Tt  = (const float*)d_in[6];
  const float* Wo  = (const float*)d_in[7];  const float* bo  = (const float*)d_in[8];
  const float* Wl1 = (const float*)d_in[9];  const float* bl1 = (const float*)d_in[10];
  const float* Wl2 = (const float*)d_in[11]; const float* bl2 = (const float*)d_in[12];
  const float* Wl3 = (const float*)d_in[13]; const float* bl3 = (const float*)d_in[14];
  const float* Wg1 = (const float*)d_in[15]; const float* as1 = (const float*)d_in[16];
  const float* ad1 = (const float*)d_in[17]; const float* bg1 = (const float*)d_in[18];
  const float* Wg2 = (const float*)d_in[19]; const float* as2 = (const float*)d_in[20];
  const float* ad2 = (const float*)d_in[21]; const float* bg2 = (const float*)d_in[22];
  const float* Wg3 = (const float*)d_in[23]; const float* as3 = (const float*)d_in[24];
  const float* ad3 = (const float*)d_in[25]; const float* bg3 = (const float*)d_in[26];
  float* outp = (float*)d_out;

  unsigned short* uw = (unsigned short*)d_ws;
  float* fw = (float*)((char*)d_ws + FBYTE);
  unsigned short* Mbf  = uw + U_MBF;
  unsigned short* TsB  = uw + U_TS;
  unsigned short* TtB  = uw + U_TT;
  unsigned short* X1B  = uw + U_X1;
  unsigned short* X23B = uw + U_X23;
  unsigned short* PbB  = uw + U_PB;
  unsigned short* WoT  = uw + U_WOT;
  unsigned short* WlT[3] = { uw + U_WL1T, uw + U_WL2T, uw + U_WL3T };
  unsigned short* WgT[3] = { uw + U_WG1T, uw + U_WG2T, uw + U_WG3T };
  float* logit = fw + F_LOG;   float* P    = fw + F_P;    float* org  = fw + F_ORG;
  float* adpP  = fw + F_ADP;   float* hbuf = fw + F_H;    float* als  = fw + F_ALS;
  float* ald   = fw + F_ALD;   unsigned* emax = (unsigned*)(fw + F_EMAX);
  float* den   = fw + F_DEN;   float* ex   = fw + F_EX;   float* gacc = fw + F_GACC;
  float* out1  = fw + F_OUT1;  float* out2 = fw + F_OUT2; float* xn   = fw + F_XN;

  auto fill = [&](unsigned* p, size_t n, unsigned v) {
    int blocks = (int)((n + 255) / 256); if (blocks > 4096) blocks = 4096;
    k_fill_u32<<<blocks, 256, 0, stream>>>(p, n, v);
  };
  auto gemm = [&](const unsigned short* A, int lda, long long sa,
                  const unsigned short* Bm, int ldb, long long sb,
                  float* C, int ldc, long long sc, int K, int M, int N, int Z, int ksplit) {
    dim3 grid(N / 64, M / 128, Z * ksplit);
    k_gemm_bf16<<<grid, 256, 0, stream>>>(A, lda, sa, Bm, ldb, sb, C, ldc, sc, K, ksplit);
  };
  auto gsblocks = [](size_t n) { int b = (int)((n + 255) / 256); return b > 8192 ? 8192 : b; };

  // 0) zero the bf16 staging prefix (guarantees all pads are 0 every launch)
  fill((unsigned*)d_ws, U_END / 2, 0u);

  // 1) pack bf16 operands
  k_cvt_pad<<<gsblocks((size_t)kN * kR), 256, 0, stream>>>(Ts, TsB, kN, kR, kRP);
  k_cvt_pad<<<gsblocks((size_t)kR * kN), 256, 0, stream>>>(Tt, TtB, kR, kN, kNP);
  k_cvt_pad<<<gsblocks((size_t)kNT * kS), 256, 0, stream>>>(T_x, X1B, kNT, kS, kS);
  k_cvt_T<<<gsblocks((size_t)kD * kS), 256, 0, stream>>>(Wo,  WoT,    kD,  kS);
  k_cvt_T<<<gsblocks((size_t)kD * kS), 256, 0, stream>>>(Wl1, WlT[0], kD,  kS);
  k_cvt_T<<<gsblocks((size_t)kD * kD), 256, 0, stream>>>(Wl2, WlT[1], kD,  kD);
  k_cvt_T<<<gsblocks((size_t)kD * kD), 256, 0, stream>>>(Wl3, WlT[2], kD,  kD);
  k_cvt_T<<<gsblocks((size_t)kHD * kS), 256, 0, stream>>>(Wg1, WgT[0], kHD, kS);
  k_cvt_T<<<gsblocks((size_t)kHD * kD), 256, 0, stream>>>(Wg2, WgT[1], kHD, kD);
  k_cvt_T<<<gsblocks((size_t)kHD * kD), 256, 0, stream>>>(Wg3, WgT[2], kHD, kD);

  // 2) M = softmax(relu(Ts @ Tt)) : WMMA GEMM [5120,128]x[128,5120] + row softmax
  gemm(TsB, kRP, 0, TtB, kNP, 0, logit, kNP, 0, kRP, kNP, kNP, 1, 1);
  k_softmax<<<kN, 256, 0, stream>>>(logit, Mbf);

  // 3) origin = x @ Wo^T (bias folded into gate1)
  gemm(X1B, kS, 0, WoT, kD, 0, org, kD, 0, kS, kNTP, kD, 1, 1);

  const float* asl[3] = { as1, as2, as3 };
  const float* adl[3] = { ad1, ad2, ad3 };
  const float* bgl[3] = { bg1, bg2, bg3 };
  const float* bll[3] = { bl1, bl2, bl3 };

  for (int l = 0; l < 3; ++l) {
    const unsigned short* Xb = (l == 0) ? X1B : X23B;
    const int Kin = (l == 0) ? kS : kD;
    if (l > 0)   // xn = tanh(out1) or relu(out2) from previous gate kernel
      k_cvt_pad<<<gsblocks((size_t)kNT * kD), 256, 0, stream>>>(xn, X23B, kNT, kD, kD);

    // adp (refactored): P = X @ Wl^T, then a = M @ P per batch, split-K x4
    gemm(Xb, Kin, 0, WlT[l], kD, 0, P, kD, 0, Kin, kNTP, kD, 1, 1);
    k_cvt_pb<<<(kNT * kD + 255) / 256, 256, 0, stream>>>(P, PbB);
    gemm(Mbf, kNP, 0, PbB, kD, (long long)kNP * kD,
         adpP, kD, (long long)kNP * kD, kNP, kNP, kD, kB, kKS);

    // GAT: h = X @ Wg^T, attention logits, seg-max/softmax over edges, scatter
    gemm(Xb, Kin, 0, WgT[l], kHD, 0, hbuf, kHD, 0, Kin, kNTP, kHD, 1, 1);
    k_attn<<<(kNT * kH + 255) / 256, 256, 0, stream>>>(hbuf, asl[l], adl[l], als, ald);
    fill(emax, (size_t)kNT * kH, 0u);
    fill((unsigned*)den, (size_t)kNT * kH, 0u);
    fill((unsigned*)gacc, (size_t)kNT * kHD, 0u);
    k_edge_max<<<(kET + 255) / 256, 256, 0, stream>>>(ei, als, ald, emax);
    k_edge_exp<<<(kET + 255) / 256, 256, 0, stream>>>(ei, als, ald, emax, ex, den);
    k_edge_scatter<<<(kET * kD + 255) / 256, 256, 0, stream>>>(ei, ex, den, hbuf, gacc);

    // gated combine
    int gb = (kNT * kD + 255) / 256;
    if (l == 0)
      k_gate1<<<gb, 256, 0, stream>>>(gacc, adpP, bgl[l], bll[l], org, bo, out1, xn);
    else if (l == 1)
      k_gate2<<<gb, 256, 0, stream>>>(gacc, adpP, bgl[l], bll[l], out1, out2, xn);
    else
      k_gate3<<<gb, 256, 0, stream>>>(gacc, adpP, bgl[l], bll[l], out2, outp);
  }
}
